// ResBlock_26173530702252
// MI455X (gfx1250) — matrix operand order repaired and checked by reading, not compiled, not run on hardware
//
#include <hip/hip_runtime.h>

typedef float  v8f   __attribute__((ext_vector_type(8)));
typedef __bf16 v16bf __attribute__((ext_vector_type(16)));

#define H_DIM     256
#define N_NODES   10000
#define E_EDGES   320000
#define NEG_SLOPE 0.2f
#define LN_EPS    1e-5f

#define WAVES            2
#define EDGES_PER_WAVE   16
#define EDGES_PER_BLOCK  (WAVES * EDGES_PER_WAVE)
#define ACT1_STRIDE      520   // 512 + 8 pad (conflict-free b128 LDS reads)
#define ACT2_STRIDE      264   // 256 + 8 pad

__device__ __forceinline__ unsigned short f2bf(float f) {
  unsigned int x = __float_as_uint(f);
  unsigned int r = x + 0x7fffu + ((x >> 16) & 1u);   // round-to-nearest-even
  return (unsigned short)(r >> 16);
}

__device__ __forceinline__ float lrelu(float x) { return x > 0.f ? x : NEG_SLOPE * x; }

__device__ __forceinline__ void atomicMaxFloat(float* addr, float v) {
  // valid for IEEE floats when slot always holds a float (init = -inf)
  if (v >= 0.f) atomicMax(reinterpret_cast<int*>(addr), __float_as_int(v));
  else          atomicMin(reinterpret_cast<unsigned int*>(addr), __float_as_uint(v));
}

// ---------------- weight pre-swizzle into WMMA B-fragment order ----------------
// Fragment (kt, nt): lane L holds 16 bf16: N = nt*16 + L%16, K = kt*32 + 8*(L/16) + e (e<8), kt*32 + 16 + 8*(L/16) + e-8 (e>=8)
__global__ void swizzle_weight_kernel(const float* __restrict__ W,
                                      unsigned short* __restrict__ out, int KT) {
  int t    = blockIdx.x * blockDim.x + threadIdx.x;
  int lane = t & 31;
  int frag = t >> 5;
  if (frag >= KT * 16) return;
  int nt = frag & 15;
  int kt = frag >> 4;
  int n  = nt * 16 + (lane & 15);
  int g  = (lane >> 4) & 1;
  unsigned short vals[16];
#pragma unroll
  for (int e = 0; e < 16; ++e) {
    int k = kt * 32 + (e < 8 ? g * 8 + e : 16 + g * 8 + (e - 8));
    vals[e] = f2bf(W[(size_t)k * H_DIM + n]);
  }
  uint4* dst = reinterpret_cast<uint4*>(out + ((size_t)frag * 32 + lane) * 16);
  dst[0] = *reinterpret_cast<uint4*>(&vals[0]);
  dst[1] = *reinterpret_cast<uint4*>(&vals[8]);
}

__global__ void init_out_kernel(float* out, int n) {
  int i = blockIdx.x * blockDim.x + threadIdx.x;
  if (i < n) out[i] = -INFINITY;
}

__global__ void finalize_kernel(float* out, const float* __restrict__ x, int n) {
  int i = blockIdx.x * blockDim.x + threadIdx.x;
  if (i < n) {
    float v = out[i];
    unsigned int b = __float_as_uint(v);
    if ((b & 0x7f800000u) == 0x7f800000u) v = 0.f;  // -inf / inf / nan -> 0
    out[i] = v + x[i];
  }
}

// ---------------- fused edge-MLP + segment-max ----------------
__global__ void __launch_bounds__(WAVES * 32, 1)
edgeconv_fused_kernel(const float* __restrict__ x,
                      const int*   __restrict__ edge_index,
                      const float* __restrict__ ew,
                      const float* __restrict__ g0, const float* __restrict__ b0,
                      const float* __restrict__ g1, const float* __restrict__ b1,
                      const float* __restrict__ g2, const float* __restrict__ b2,
                      const unsigned short* __restrict__ w1s,
                      const unsigned short* __restrict__ w2s,
                      const unsigned short* __restrict__ w3s,
                      float* __restrict__ out) {
  __shared__ __align__(16) unsigned short act[WAVES][16 * ACT1_STRIDE];

  const int tid   = threadIdx.x;
  const int lane  = tid & 31;
  const int wave  = tid >> 5;
  const int g     = (lane >> 4) & 1;
  const int ln16  = lane & 15;
  unsigned short* actw = act[wave];
  const int ebase = blockIdx.x * EDGES_PER_BLOCK + wave * EDGES_PER_WAVE;
  const int* srcp = edge_index;
  const int* dstp = edge_index + E_EDGES;

  // ---- Stage 0: gather + LN0 + LReLU -> LDS bf16 [16][512]
  for (int el = 0; el < EDGES_PER_WAVE; ++el) {
    int e = ebase + el;
    int sj = srcp[e], di = dstp[e];
    float w = ew[e];
    const float* xi = x + (size_t)di * H_DIM;
    const float* xj = x + (size_t)sj * H_DIM;
    float lo[8], hi[8];
    float s = 0.f, ss = 0.f;
#pragma unroll
    for (int k = 0; k < 8; ++k) {
      int f = lane + 32 * k;
      float a = xi[f];
      float b = w * (xj[f] - a);
      lo[k] = a; hi[k] = b;
      s += a + b; ss += a * a + b * b;
    }
#pragma unroll
    for (int off = 16; off >= 1; off >>= 1) {
      s  += __shfl_xor(s,  off, 32);
      ss += __shfl_xor(ss, off, 32);
    }
    float mean = s * (1.f / 512.f);
    float rstd = rsqrtf(ss * (1.f / 512.f) - mean * mean + LN_EPS);
    unsigned short* row = actw + el * ACT1_STRIDE;
#pragma unroll
    for (int k = 0; k < 8; ++k) {
      int f = lane + 32 * k;
      row[f]       = f2bf(lrelu((lo[k] - mean) * rstd * g0[f]       + b0[f]));
      row[256 + f] = f2bf(lrelu((hi[k] - mean) * rstd * g0[256 + f] + b0[256 + f]));
    }
  }

  v8f acc[16];
  const v8f vzero = {0.f, 0.f, 0.f, 0.f, 0.f, 0.f, 0.f, 0.f};

  // ---- GEMM1: [16x512] @ W1[512x256]
#pragma unroll
  for (int nt = 0; nt < 16; ++nt) acc[nt] = vzero;
#pragma unroll 1
  for (int kt = 0; kt < 16; ++kt) {
    union { uint4 q[2]; v16bf v; } A;
    const unsigned short* ap = actw + ln16 * ACT1_STRIDE + kt * 32 + 8 * g;
    A.q[0] = *reinterpret_cast<const uint4*>(ap);
    A.q[1] = *reinterpret_cast<const uint4*>(ap + 16);
#pragma unroll
    for (int nt = 0; nt < 16; ++nt) {
      union { uint4 q[2]; v16bf v; } B;
      const uint4* bp = reinterpret_cast<const uint4*>(w1s) + ((size_t)(kt * 16 + nt) * 32 + lane) * 2;
      B.q[0] = bp[0]; B.q[1] = bp[1];
      acc[nt] = __builtin_amdgcn_wmma_f32_16x16x32_bf16(false, A.v, false, B.v,
                                                        (short)0, acc[nt], false, false);
    }
  }

  // ---- LN1 + LReLU (in-register stats) -> LDS bf16 [16][256]
#pragma unroll 1
  for (int r = 0; r < 8; ++r) {
    float s = 0.f, ss = 0.f;
#pragma unroll
    for (int nt = 0; nt < 16; ++nt) { float v = acc[nt][r]; s += v; ss += v * v; }
#pragma unroll
    for (int off = 8; off >= 1; off >>= 1) {
      s  += __shfl_xor(s,  off, 32);
      ss += __shfl_xor(ss, off, 32);
    }
    float mean = s * (1.f / 256.f);
    float rstd = rsqrtf(ss * (1.f / 256.f) - mean * mean + LN_EPS);
    unsigned short* rowp = actw + (r + 8 * g) * ACT2_STRIDE;
#pragma unroll
    for (int nt = 0; nt < 16; ++nt) {
      int f = nt * 16 + ln16;
      rowp[f] = f2bf(lrelu((acc[nt][r] - mean) * rstd * g1[f] + b1[f]));
    }
  }

  // ---- GEMM2: [16x256] @ W2[256x256]
#pragma unroll
  for (int nt = 0; nt < 16; ++nt) acc[nt] = vzero;
#pragma unroll 1
  for (int kt = 0; kt < 8; ++kt) {
    union { uint4 q[2]; v16bf v; } A;
    const unsigned short* ap = actw + ln16 * ACT2_STRIDE + kt * 32 + 8 * g;
    A.q[0] = *reinterpret_cast<const uint4*>(ap);
    A.q[1] = *reinterpret_cast<const uint4*>(ap + 16);
#pragma unroll
    for (int nt = 0; nt < 16; ++nt) {
      union { uint4 q[2]; v16bf v; } B;
      const uint4* bp = reinterpret_cast<const uint4*>(w2s) + ((size_t)(kt * 16 + nt) * 32 + lane) * 2;
      B.q[0] = bp[0]; B.q[1] = bp[1];
      acc[nt] = __builtin_amdgcn_wmma_f32_16x16x32_bf16(false, A.v, false, B.v,
                                                        (short)0, acc[nt], false, false);
    }
  }

  // ---- LN2 + LReLU -> LDS bf16 [16][256]
#pragma unroll 1
  for (int r = 0; r < 8; ++r) {
    float s = 0.f, ss = 0.f;
#pragma unroll
    for (int nt = 0; nt < 16; ++nt) { float v = acc[nt][r]; s += v; ss += v * v; }
#pragma unroll
    for (int off = 8; off >= 1; off >>= 1) {
      s  += __shfl_xor(s,  off, 32);
      ss += __shfl_xor(ss, off, 32);
    }
    float mean = s * (1.f / 256.f);
    float rstd = rsqrtf(ss * (1.f / 256.f) - mean * mean + LN_EPS);
    unsigned short* rowp = actw + (r + 8 * g) * ACT2_STRIDE;
#pragma unroll
    for (int nt = 0; nt < 16; ++nt) {
      int f = nt * 16 + ln16;
      rowp[f] = f2bf(lrelu((acc[nt][r] - mean) * rstd * g2[f] + b2[f]));
    }
  }

  // ---- GEMM3: [16x256] @ W3[256x256]
#pragma unroll
  for (int nt = 0; nt < 16; ++nt) acc[nt] = vzero;
#pragma unroll 1
  for (int kt = 0; kt < 8; ++kt) {
    union { uint4 q[2]; v16bf v; } A;
    const unsigned short* ap = actw + ln16 * ACT2_STRIDE + kt * 32 + 8 * g;
    A.q[0] = *reinterpret_cast<const uint4*>(ap);
    A.q[1] = *reinterpret_cast<const uint4*>(ap + 16);
#pragma unroll
    for (int nt = 0; nt < 16; ++nt) {
      union { uint4 q[2]; v16bf v; } B;
      const uint4* bp = reinterpret_cast<const uint4*>(w3s) + ((size_t)(kt * 16 + nt) * 32 + lane) * 2;
      B.q[0] = bp[0]; B.q[1] = bp[1];
      acc[nt] = __builtin_amdgcn_wmma_f32_16x16x32_bf16(false, A.v, false, B.v,
                                                        (short)0, acc[nt], false, false);
    }
  }

  // ---- segment-max scatter: atomic max into out[dst]
#pragma unroll 1
  for (int r = 0; r < 8; ++r) {
    int m = r + 8 * g;
    int e = ebase + m;
    int di = dstp[e];
    float* orow = out + (size_t)di * H_DIM;
#pragma unroll
    for (int nt = 0; nt < 16; ++nt) {
      atomicMaxFloat(orow + nt * 16 + ln16, acc[nt][r]);
    }
  }
}

extern "C" void kernel_launch(void* const* d_in, const int* in_sizes, int n_in,
                              void* d_out, int out_size, void* d_ws, size_t ws_size,
                              hipStream_t stream) {
  const float* x  = (const float*)d_in[0];
  const int*   ei = (const int*)d_in[1];
  const float* ew = (const float*)d_in[2];
  const float* g0 = (const float*)d_in[3];
  const float* b0 = (const float*)d_in[4];
  const float* W1 = (const float*)d_in[5];
  const float* g1 = (const float*)d_in[6];
  const float* b1 = (const float*)d_in[7];
  const float* W2 = (const float*)d_in[8];
  const float* g2 = (const float*)d_in[9];
  const float* b2 = (const float*)d_in[10];
  const float* W3 = (const float*)d_in[11];
  float* out = (float*)d_out;

  unsigned short* w1s = (unsigned short*)d_ws;        // 512*256 bf16
  unsigned short* w2s = w1s + 512 * 256;              // 256*256 bf16
  unsigned short* w3s = w2s + 256 * 256;              // 256*256 bf16

  swizzle_weight_kernel<<<(16 * 16 * 32 + 255) / 256, 256, 0, stream>>>(W1, w1s, 16);
  swizzle_weight_kernel<<<( 8 * 16 * 32 + 255) / 256, 256, 0, stream>>>(W2, w2s, 8);
  swizzle_weight_kernel<<<( 8 * 16 * 32 + 255) / 256, 256, 0, stream>>>(W3, w3s, 8);

  init_out_kernel<<<(N_NODES * H_DIM + 255) / 256, 256, 0, stream>>>(out, N_NODES * H_DIM);

  edgeconv_fused_kernel<<<E_EDGES / EDGES_PER_BLOCK, WAVES * 32, 0, stream>>>(
      x, ei, ew, g0, b0, g1, b1, g2, b2, w1s, w2s, w3s, out);

  finalize_kernel<<<(N_NODES * H_DIM + 255) / 256, 256, 0, stream>>>(out, x, N_NODES * H_DIM);
}
